// ESNBase_10625749090431
// MI455X (gfx1250) — matrix operand order repaired and checked by reading, not compiled, not run on hardware
//
#include <hip/hip_runtime.h>

// ---------------------------------------------------------------------------
// CDNA5 (gfx1250) ESN: single persistent-kernel scan.
// Per step/layer: 64x2048 = h @ W^T (+ x @ Win^T) via V_WMMA_F32_16X16X32_BF16,
// fp32 accumulate, fused leaky-tanh update. 128 waves, each owns a 64x16 strip
// (4 M-tiles x 1 N-tile) so every weight fragment feeds 4 WMMAs and the R x R
// weight matrices are streamed from L2 exactly once per step.
// ---------------------------------------------------------------------------

typedef __attribute__((ext_vector_type(16))) __bf16 v16bf;
typedef __attribute__((ext_vector_type(8)))  float  v8f;

union frag16 {
    v16bf v;
    uint4 u[2];
};

static __device__ __forceinline__ unsigned short f32_to_bf16(float f) {
    union { float f; unsigned int u; } v;
    v.f = f;
    unsigned int lsb = (v.u >> 16) & 1u;
    v.u += 0x7FFFu + lsb;               // round-to-nearest-even
    return (unsigned short)(v.u >> 16);
}

// ------------------------- elementwise helpers ------------------------------

__global__ void cvt_f32_bf16_kernel(const float* __restrict__ src,
                                    unsigned short* __restrict__ dst, int n) {
    int i = blockIdx.x * blockDim.x + threadIdx.x;
    if (i < n) dst[i] = f32_to_bf16(src[i]);
}

__global__ void zero_u32_kernel(unsigned int* __restrict__ p, int n) {
    int i = blockIdx.x * blockDim.x + threadIdx.x;
    if (i < n) p[i] = 0u;
}

// ------------------------- grid-wide barrier --------------------------------
// Generation barrier over all resident blocks (16 blocks -> always resident).

static __device__ __forceinline__ void grid_sync(unsigned int* count,
                                                 unsigned int* gen,
                                                 unsigned int nblk) {
    __threadfence();                 // make this thread's stores device-visible
    __syncthreads();
    if (threadIdx.x == 0) {
        unsigned int g = *(volatile unsigned int*)gen;
        unsigned int arrived = atomicAdd(count, 1u) + 1u;
        if (arrived == nblk) {
            atomicExch(count, 0u);   // reset before releasing
            __threadfence();
            atomicAdd(gen, 1u);
        } else {
            while (*(volatile unsigned int*)gen == g) {
                __builtin_amdgcn_s_sleep(1);
            }
        }
    }
    __syncthreads();
    __threadfence();                 // acquire other blocks' stores
}

// ------------------------- WMMA strip GEMM ----------------------------------
// acc[m][b][n] += sum_k A[16m+b][k] * W[n0+n][k], m = 0..3 (whole 64-row batch)
// Per-lane operand loads are contiguous 16B/32B vectors in the documented
// WMMA register layouts (CDNA5 ISA §7.12.2).

static __device__ __forceinline__ void gemm_strip(v8f* acc,
                                                  const __bf16* __restrict__ A,
                                                  const __bf16* __restrict__ W,
                                                  int K, int n0, int lm, int lh) {
    const __bf16* brow = W + (size_t)(n0 + lm) * K + lh * 8;
    const __bf16* arow = A + (size_t)lm * K + lh * 8;
    const size_t mstride = (size_t)16 * K;
    for (int k = 0; k < K; k += 32) {
        frag16 b;
        b.u[0] = *(const uint4*)(brow + k);        // K = lh*8 + 0..7
        b.u[1] = *(const uint4*)(brow + k + 16);   // K = 16 + lh*8 + 0..7
#pragma unroll
        for (int m = 0; m < 4; ++m) {
            frag16 a;
            a.u[0] = *(const uint4*)(arow + m * mstride + k);       // K=lh*8+0..7
            a.u[1] = *(const uint4*)(arow + m * mstride + k + 16);  // +16
            acc[m] = __builtin_amdgcn_wmma_f32_16x16x32_bf16(
                false, a.v, false, b.v, (short)0, acc[m], false, false);
        }
    }
}

// h' = 0.1*h + 0.9*tanh(acc + bias); update fp32 state, bf16 mirror, optional y.
static __device__ __forceinline__ void strip_epilogue(v8f* acc, float bv,
                                                      int n0, int lm, int lh,
                                                      float* __restrict__ h_state,
                                                      unsigned short* __restrict__ h_bf_out,
                                                      float* __restrict__ y_out) {
#pragma unroll
    for (int m = 0; m < 4; ++m) {
#pragma unroll
        for (int v = 0; v < 8; ++v) {
            const int b   = m * 16 + v + 8 * lh;   // C/D: M = v (+8 for lanes 16-31)
            const int idx = b * 2048 + n0 + lm;    // N = lane & 15
            const float pre  = acc[m][v] + bv;
            const float hn   = 0.1f * h_state[idx] + 0.9f * tanhf(pre);
            h_state[idx]  = hn;
            h_bf_out[idx] = f32_to_bf16(hn);
            if (y_out) y_out[idx] = hn;
        }
    }
}

// ------------------------- persistent scan kernel ---------------------------
// Grid: 16 blocks x 256 threads = 128 waves; wave g owns reservoir cols
// [16g, 16g+16). Two grid barriers per timestep (after layer 0, after layer 1).

__global__ __launch_bounds__(256)
void esn_scan_kernel(const __bf16* __restrict__ x_bf,
                     const __bf16* __restrict__ Win0, const __bf16* __restrict__ Whh0,
                     const __bf16* __restrict__ Win1, const __bf16* __restrict__ Whh1,
                     const float* __restrict__ b0, const float* __restrict__ b1,
                     float* __restrict__ h0_f32, float* __restrict__ h1_f32,
                     unsigned short* __restrict__ h0_bf0, unsigned short* __restrict__ h0_bf1,
                     unsigned short* __restrict__ h1_bf0, unsigned short* __restrict__ h1_bf1,
                     float* __restrict__ y_out, unsigned int* __restrict__ bar) {
    const int S = 512, B = 64, I = 256;
    const int lane = threadIdx.x & 31;
    const int wave = threadIdx.x >> 5;
    const int lm = lane & 15;
    const int lh = lane >> 4;
    const int n0 = (blockIdx.x * 8 + wave) << 4;   // 0..2032

    const float bv0 = b0[n0 + lm];
    const float bv1 = b1[n0 + lm];

    unsigned short* h0bf[2] = { h0_bf0, h0_bf1 };
    unsigned short* h1bf[2] = { h1_bf0, h1_bf1 };
    const unsigned int nblk = gridDim.x;

    for (int t = 0; t < S; ++t) {
        const int cur = t & 1, nxt = cur ^ 1;

        // ---- layer 0: acc = x_t @ Win0^T + h0 @ Whh0^T ----
        {
            v8f acc[4] = {};
            gemm_strip(acc, x_bf + (size_t)t * B * I, Win0, I,    n0, lm, lh);
            gemm_strip(acc, (const __bf16*)h0bf[cur], Whh0, 2048, n0, lm, lh);
            strip_epilogue(acc, bv0, n0, lm, lh, h0_f32, h0bf[nxt], nullptr);
        }
        grid_sync(bar, bar + 1, nblk);

        // ---- layer 1: acc = y0_t @ Win1^T + h1 @ Whh1^T  (y0_t == h0bf[nxt]) ----
        {
            v8f acc[4] = {};
            gemm_strip(acc, (const __bf16*)h0bf[nxt], Win1, 2048, n0, lm, lh);
            gemm_strip(acc, (const __bf16*)h1bf[cur], Whh1, 2048, n0, lm, lh);
            strip_epilogue(acc, bv1, n0, lm, lh, h1_f32, h1bf[nxt],
                           y_out + (size_t)t * B * 2048);
        }
        grid_sync(bar, bar + 1, nblk);
    }
}

// ---------------------------------------------------------------------------

extern "C" void kernel_launch(void* const* d_in, const int* in_sizes, int n_in,
                              void* d_out, int out_size, void* d_ws, size_t ws_size,
                              hipStream_t stream) {
    const int S = 512, B = 64, I = 256, R = 2048;

    const float* x     = (const float*)d_in[0];
    const float* W_in0 = (const float*)d_in[1];
    const float* W_hh0 = (const float*)d_in[2];
    const float* b0    = (const float*)d_in[3];
    const float* W_in1 = (const float*)d_in[4];
    const float* W_hh1 = (const float*)d_in[5];
    const float* b1    = (const float*)d_in[6];
    float* out = (float*)d_out;

    // -------- workspace carve-up (256B aligned) --------
    char* w = (char*)d_ws;
    auto carve = [&](size_t bytes) -> char* {
        char* p = w;
        w += (bytes + 255) & ~(size_t)255;
        return p;
    };
    unsigned short* x_bf    = (unsigned short*)carve((size_t)S * B * I * 2);
    unsigned short* Win0_bf = (unsigned short*)carve((size_t)R * I * 2);
    unsigned short* Whh0_bf = (unsigned short*)carve((size_t)R * R * 2);
    unsigned short* Win1_bf = (unsigned short*)carve((size_t)R * R * 2);
    unsigned short* Whh1_bf = (unsigned short*)carve((size_t)R * R * 2);
    // state block (zero-initialized every call, contiguous):
    char* state_base = w;
    float*          h0_f32 = (float*)carve((size_t)B * R * 4);
    float*          h1_f32 = (float*)carve((size_t)B * R * 4);
    unsigned short* h0_bf0 = (unsigned short*)carve((size_t)B * R * 2);
    unsigned short* h0_bf1 = (unsigned short*)carve((size_t)B * R * 2);
    unsigned short* h1_bf0 = (unsigned short*)carve((size_t)B * R * 2);
    unsigned short* h1_bf1 = (unsigned short*)carve((size_t)B * R * 2);
    unsigned int*   bar    = (unsigned int*)carve(256);   // {count, gen}
    const size_t state_bytes = (size_t)(w - state_base);

    // -------- 1) convert inputs/weights to bf16 --------
    auto cvt = [&](const float* src, unsigned short* dst, int n) {
        cvt_f32_bf16_kernel<<<(n + 255) / 256, 256, 0, stream>>>(src, dst, n);
    };
    cvt(x,     x_bf,    S * B * I);
    cvt(W_in0, Win0_bf, R * I);
    cvt(W_hh0, Whh0_bf, R * R);
    cvt(W_in1, Win1_bf, R * R);
    cvt(W_hh1, Whh1_bf, R * R);

    // -------- 2) zero state + barrier block --------
    {
        int n = (int)(state_bytes / 4);
        zero_u32_kernel<<<(n + 255) / 256, 256, 0, stream>>>(
            (unsigned int*)state_base, n);
    }

    // -------- 3) one persistent kernel runs the whole 512-step scan --------
    esn_scan_kernel<<<16, 256, 0, stream>>>(
        (const __bf16*)x_bf,
        (const __bf16*)Win0_bf, (const __bf16*)Whh0_bf,
        (const __bf16*)Win1_bf, (const __bf16*)Whh1_bf,
        b0, b1, h0_f32, h1_f32,
        h0_bf0, h0_bf1, h1_bf0, h1_bf1,
        out, bar);

    // -------- 4) final hidden states -> d_out tail: h_n = [h0_n, h1_n] --------
    hipMemcpyAsync(out + (size_t)S * B * R,
                   h0_f32, (size_t)B * R * sizeof(float),
                   hipMemcpyDeviceToDevice, stream);
    hipMemcpyAsync(out + (size_t)S * B * R + (size_t)B * R,
                   h1_f32, (size_t)B * R * sizeof(float),
                   hipMemcpyDeviceToDevice, stream);
}
